// Attention_49211735277969
// MI455X (gfx1250) — matrix operand order repaired and checked by reading, not compiled, not run on hardware
//
#include <hip/hip_runtime.h>
#include <hip/hip_bf16.h>
#include <stdint.h>

#ifndef __has_builtin
#define __has_builtin(x) 0
#endif

// CDNA5 async global->LDS DMA path (probe-confirmed on this toolchain).
#if __has_builtin(__builtin_amdgcn_global_load_async_to_lds_b128)
#define HAVE_ASYNC_LDS 1
#else
#define HAVE_ASYNC_LDS 0
#endif

#define GLOBAL_AS __attribute__((address_space(1)))
#define LDS_AS    __attribute__((address_space(3)))

// ---------------------------------------------------------------------------
// CDNA5 WMMA types / fragment helpers (wave32, v_wmma_f32_16x16x32_f16)
// ---------------------------------------------------------------------------
typedef __attribute__((ext_vector_type(16))) _Float16 v16h;
typedef __attribute__((ext_vector_type(8)))  float    v8f;
typedef __attribute__((ext_vector_type(4)))  int      v4i;

union FragH { v16h v; uint32_t u[8]; };

// A-fragment (16x32 f16, MxK), ISA 7.12.2 layout.
__device__ __forceinline__ v16h load_afrag(const _Float16* s, int rowBase,
                                           int stride, int kOff, int lane) {
  FragH f;
  const _Float16* row = s + (size_t)(rowBase + (lane & 15)) * stride;
  int kh = (lane >> 4) * 8;
#pragma unroll
  for (int v = 0; v < 8; ++v) {
    int k = kOff + kh + (v < 4 ? 2 * v : 16 + 2 * (v - 4));
    f.u[v] = *(const uint32_t*)(row + k);
  }
  return f.v;
}

// B-fragment (32x16 f16, KxN), data stored [n][k] row-major over k.
__device__ __forceinline__ v16h load_bfrag(const _Float16* s, int colBase,
                                           int stride, int kOff, int lane) {
  FragH f;
  const _Float16* row = s + (size_t)(colBase + (lane & 15)) * stride;
  int kb = kOff + (lane >> 4) * 8;
#pragma unroll
  for (int v = 0; v < 8; ++v) f.u[v] = *(const uint32_t*)(row + kb + (v < 4 ? 2 * v : 16 + 2 * (v - 4)));
  return f.v;
}

// 16-byte global -> LDS copy: async DMA when available.
__device__ __forceinline__ void copy_b128(const void* gsrc, void* ldst) {
#if HAVE_ASYNC_LDS
  __builtin_amdgcn_global_load_async_to_lds_b128(
      (GLOBAL_AS v4i*)gsrc, (LDS_AS v4i*)ldst, 0, 0);
#else
  *(uint4*)ldst = *(const uint4*)gsrc;
#endif
}

__device__ __forceinline__ void wait_async_copies() {
#if HAVE_ASYNC_LDS
#if __has_builtin(__builtin_amdgcn_s_wait_asynccnt)
  __builtin_amdgcn_s_wait_asynccnt(0);
#else
  asm volatile("s_wait_asynccnt 0" ::: "memory");
#endif
#endif
}

// ---------------------------------------------------------------------------
// Elementwise f32 -> f16 conversion (vectorized x4)
// ---------------------------------------------------------------------------
__global__ void cvt_f32_f16(const float* __restrict__ src,
                            _Float16* __restrict__ dst, size_t n4) {
  size_t i = (size_t)blockIdx.x * blockDim.x + threadIdx.x;
  if (i >= n4) return;
  float4 f = ((const float4*)src)[i];
  union { _Float16 h[4]; uint64_t q; } o;
  o.h[0] = (_Float16)f.x; o.h[1] = (_Float16)f.y;
  o.h[2] = (_Float16)f.z; o.h[3] = (_Float16)f.w;
  *(uint64_t*)(dst + i * 4) = o.q;
}

// ---------------------------------------------------------------------------
// Double-buffered WMMA GEMM:  C[M,N] = A[M,K] @ Bt[N,K]^T + bias[N]
// A,Bt f16 row-major, C f32.  Block 128x128, K-step 32, 8 waves (4x2),
// each wave 32x64 = 2x4 WMMA tiles.  Async B128 global->LDS staging with
// flat (base + bsel*BUFSZ + thread-offset) addressing -> no pointer selects.
// ---------------------------------------------------------------------------
#define BM 128
#define BN 128
#define BK 32
#define SAS 40   // padded LDS K-stride (halves): 80 B rows -> 16B-aligned b128
                 // stores, 20-dword lane stride -> conflict-free frag reads
#define GBUF (BM * SAS)   // halves per buffer

__global__ __launch_bounds__(256) void gemm_f16f32(
    const _Float16* __restrict__ A, const _Float16* __restrict__ Bt,
    const float* __restrict__ bias, float* __restrict__ C,
    int M, int N, int K) {
  __shared__ _Float16 sA[2 * GBUF];
  __shared__ _Float16 sB[2 * GBUF];

  const int nb   = N / BN;
  const int bm   = blockIdx.x / nb, bn = blockIdx.x % nb;
  const int tid  = threadIdx.x;
  const int wave = tid >> 5, lane = tid & 31;
  const int wm   = wave >> 1, wn = wave & 1;   // 4 (M) x 2 (N) wave grid
  const int m0   = bm * BM, n0 = bn * BN;

  // Per-thread staging coords: 512 b128 chunks per 128x32 tile,
  // 256 threads x 2 chunks each (per matrix).
  int lOff[2];
  const _Float16 *aG[2], *bG[2];
#pragma unroll
  for (int j = 0; j < 2; ++j) {
    int idx = tid + j * 256;
    int r = idx >> 2, c = (idx & 3) * 8;
    lOff[j] = r * SAS + c;
    aG[j] = A  + (size_t)(m0 + r) * K + c;
    bG[j] = Bt + (size_t)(n0 + r) * K + c;
  }

  v8f acc[2][4] = {};

  const int nt = K / BK;
  // Prologue: stage tile 0 into buffer 0.
#pragma unroll
  for (int j = 0; j < 2; ++j) {
    copy_b128(aG[j], &sA[lOff[j]]);
    copy_b128(bG[j], &sB[lOff[j]]);
  }

  for (int it = 0; it < nt; ++it) {
    const int curOff = (it & 1) * GBUF;        // uniform scalar offset
    wait_async_copies();
    __syncthreads();                           // tile `it` visible

    if (it + 1 < nt) {                         // stage tile it+1 into other buf
      const int nxtOff = curOff ^ GBUF;
      const int k0n = (it + 1) * BK;
#pragma unroll
      for (int j = 0; j < 2; ++j) {
        copy_b128(aG[j] + k0n, &sA[nxtOff + lOff[j]]);
        copy_b128(bG[j] + k0n, &sB[nxtOff + lOff[j]]);
      }
    }

    v16h af[2], bf[4];
#pragma unroll
    for (int i = 0; i < 2; ++i)
      af[i] = load_afrag(&sA[curOff], wm * 32 + i * 16, SAS, 0, lane);
#pragma unroll
    for (int j = 0; j < 4; ++j)
      bf[j] = load_bfrag(&sB[curOff], wn * 64 + j * 16, SAS, 0, lane);

#pragma unroll
    for (int i = 0; i < 2; ++i)
#pragma unroll
      for (int j = 0; j < 4; ++j)
        acc[i][j] = __builtin_amdgcn_wmma_f32_16x16x32_f16(
            false, af[i], false, bf[j], (short)0, acc[i][j], false, false);
  }

  // Epilogue: C-layout: VGPR r holds row r + 8*(lane>>4), col lane&15.
  const int half = lane >> 4, nl = lane & 15;
#pragma unroll
  for (int i = 0; i < 2; ++i)
#pragma unroll
    for (int j = 0; j < 4; ++j) {
      int n = n0 + wn * 64 + j * 16 + nl;
      float bv = bias ? bias[n] : 0.0f;
#pragma unroll
      for (int r = 0; r < 8; ++r) {
        int m = m0 + wm * 32 + i * 16 + r + 8 * half;
        C[(size_t)m * N + n] = acc[i][j][r] + bv;
      }
    }
}

// ---------------------------------------------------------------------------
// RoPE + repack to head-major f16:  proj [B*S][nheads*64] -> out [B][H][S][64]
// ---------------------------------------------------------------------------
__global__ void rope_pack(const float* __restrict__ proj,
                          const float* __restrict__ freqs,
                          _Float16* __restrict__ outH,
                          int nheads, int width, size_t npairs) {
  size_t idx = (size_t)blockIdx.x * blockDim.x + threadIdx.x;
  if (idx >= npairs) return;
  int pairsPerRow = width >> 1;
  int row  = (int)(idx / pairsPerRow);   // b*S + s
  int p    = (int)(idx % pairsPerRow);
  int head = p >> 5, i = p & 31;         // 32 pairs per 64-dim head
  int b    = row >> 11, s = row & 2047;  // S = 2048
  float c  = freqs[s * 64 + 2 * i];
  float sn = freqs[s * 64 + 2 * i + 1];
  float xe = proj[(size_t)row * width + head * 64 + 2 * i];
  float xo = proj[(size_t)row * width + head * 64 + 2 * i + 1];
  float re = xe * c - xo * sn;
  float ro = xe * sn + xo * c;
  size_t o = (((size_t)(b * nheads + head)) * 2048 + s) * 64 + 2 * i;
  union { _Float16 h[2]; uint32_t u; } pk;
  pk.h[0] = (_Float16)re; pk.h[1] = (_Float16)ro;
  *(uint32_t*)(outH + o) = pk.u;
}

// V: convert + repack head-major (no RoPE). vf [B*S][512] -> vh [B][8][S][64]
__global__ void v_pack(const float* __restrict__ vf, _Float16* __restrict__ vh,
                       size_t n) {
  size_t idx = (size_t)blockIdx.x * blockDim.x + threadIdx.x;
  if (idx >= n) return;
  int row = (int)(idx >> 9), col = (int)(idx & 511);
  int g = col >> 6, d = col & 63;
  int b = row >> 11, s = row & 2047;
  vh[(((size_t)(b * 8 + g)) * 2048 + s) * 64 + d] = (_Float16)vf[idx];
}

// ---------------------------------------------------------------------------
// Flash attention (causal, GQA rep=4).  One WG = (b, head, 128 query rows),
// 8 waves x 16 rows each.  WMMA for QK^T and P@V, f32 online softmax.
// ---------------------------------------------------------------------------
#define SKD 72   // K tile [t][d] stride: 144 B rows (16B aligned), 36-dword
                 // lane stride -> conflict-free b-frag reads
#define SVD 34   // V^T tile [d][t] padded stride
#define SPD 34   // P tile [m][t] padded stride

__global__ __launch_bounds__(256) void attn_fwd(
    const _Float16* __restrict__ Qh, const _Float16* __restrict__ Kh,
    const _Float16* __restrict__ Vh, _Float16* __restrict__ Oh) {
  __shared__ _Float16 sK[32 * SKD];
  __shared__ _Float16 sV[64 * SVD];
  __shared__ _Float16 sP[8 * 16 * SPD];

  const int bid = blockIdx.x;
  const int qb  = (bid & 15) * 128;       // 2048/128 = 16 q-blocks
  const int bh  = bid >> 4;               // b*32 + h
  const int h   = bh & 31, b = bh >> 5;
  const int g   = h >> 2;                 // GQA: 32 heads / 8 groups

  const int tid = threadIdx.x, wave = tid >> 5, lane = tid & 31;
  const int half = lane >> 4, nl = lane & 15;
  const int qm0 = qb + wave * 16;

  // This wave's Q rows, loaded straight into A-fragment layout (2 d-chunks).
  FragH qa[2];
  {
    const _Float16* qrow = Qh + (((size_t)bh) * 2048 + qm0 + nl) * 64;
#pragma unroll
    for (int kc = 0; kc < 2; ++kc)
#pragma unroll
      for (int v = 0; v < 8; ++v) {
        int k = kc * 32 + half * 8 + (v < 4 ? 2 * v : 16 + 2 * (v - 4));
        qa[kc].u[v] = *(const uint32_t*)(qrow + k);
      }
  }

  const _Float16* kbase = Kh + ((size_t)(b * 8 + g)) * 2048 * 64;
  const _Float16* vbase = Vh + ((size_t)(b * 8 + g)) * 2048 * 64;

  // Per-thread staging coords: K tile = 32x64 halves = 256 b128 chunks.
  const int kt = tid >> 3, kc8 = (tid & 7) * 8;
  const _Float16* kG = kbase + (size_t)kt * 64 + kc8;
  _Float16*       kL = &sK[kt * SKD + kc8];

  float mrun[8], lrun[8];
  v8f o[4] = {};
#pragma unroll
  for (int r = 0; r < 8; ++r) { mrun[r] = -3.0e38f; lrun[r] = 0.0f; }

  const int tend = qb + 128;  // causal: no key beyond the block diagonal

  for (int t0 = 0; t0 < tend; t0 += 32) {
    __syncthreads();
    // K tile via async b128 copies; V tile transposed with scalar stores.
    copy_b128(kG + (size_t)t0 * 64, kL);
#pragma unroll
    for (int j = 0; j < 4; ++j) {
      int idx = tid + j * 256;              // 1024 dwords
      int t = idx >> 5, d2 = idx & 31;
      union { uint32_t u; _Float16 f[2]; } cv;
      cv.u = *(const uint32_t*)&vbase[(size_t)(t0 + t) * 64 + d2 * 2];
      sV[(d2 * 2) * SVD + t]     = cv.f[0];
      sV[(d2 * 2 + 1) * SVD + t] = cv.f[1];
    }
    wait_async_copies();
    __syncthreads();

    // scores S = Q K^T  (two 16-col tiles, d-dim 64 in two 32 chunks)
    v8f sc[2] = {};
#pragma unroll
    for (int nt = 0; nt < 2; ++nt)
#pragma unroll
      for (int kc = 0; kc < 2; ++kc) {
        v16h bk = load_bfrag(sK, nt * 16, SKD, kc * 32, lane);
        sc[nt] = __builtin_amdgcn_wmma_f32_16x16x32_f16(
            false, qa[kc].v, false, bk, (short)0, sc[nt], false, false);
      }

    // online softmax per row (row = r + 8*half; cols striped over lanes 0..15)
#pragma unroll
    for (int r = 0; r < 8; ++r) {
      int qm = qm0 + r + 8 * half;
      float mx = -3.0e38f;
#pragma unroll
      for (int nt = 0; nt < 2; ++nt) {
        float s = sc[nt][r] * 0.125f;       // 1/sqrt(64)
        int tg = t0 + nt * 16 + nl;
        if (tg > qm) s = -1.0e9f;           // causal mask
        sc[nt][r] = s;
        mx = fmaxf(mx, s);
      }
#pragma unroll
      for (int m = 1; m < 16; m <<= 1) mx = fmaxf(mx, __shfl_xor(mx, m, 32));
      float mnew = fmaxf(mrun[r], mx);
      float rsum = 0.0f;
#pragma unroll
      for (int nt = 0; nt < 2; ++nt) {
        float p = __expf(sc[nt][r] - mnew);
        sc[nt][r] = p;
        rsum += p;
      }
#pragma unroll
      for (int m = 1; m < 16; m <<= 1) rsum += __shfl_xor(rsum, m, 32);
      float sc_old = __expf(mrun[r] - mnew);
      lrun[r] = lrun[r] * sc_old + rsum;
      mrun[r] = mnew;
#pragma unroll
      for (int dn = 0; dn < 4; ++dn) o[dn][r] *= sc_old;

      // spill P row (f16) to wave-private LDS: C-layout -> A-layout reshape
      _Float16* prow = &sP[(size_t)(wave * 16 + r + 8 * half) * SPD];
      prow[nl]      = (_Float16)sc[0][r];
      prow[16 + nl] = (_Float16)sc[1][r];
    }

    asm volatile("s_wait_dscnt 0" ::: "memory");  // wave-local LDS RAW fence

    v16h pa = load_afrag(&sP[(size_t)wave * 16 * SPD], 0, SPD, 0, lane);
#pragma unroll
    for (int dn = 0; dn < 4; ++dn) {
      v16h bv = load_bfrag(sV, dn * 16, SVD, 0, lane);
      o[dn] = __builtin_amdgcn_wmma_f32_16x16x32_f16(
          false, pa, false, bv, (short)0, o[dn], false, false);
    }
  }

  // Normalize and store output repacked to [b][s][h*64+d] (f16)
#pragma unroll
  for (int dn = 0; dn < 4; ++dn)
#pragma unroll
    for (int r = 0; r < 8; ++r) {
      int qm = qm0 + r + 8 * half;
      int d  = dn * 16 + nl;
      Oh[((size_t)b * 2048 + qm) * 2048 + h * 64 + d] =
          (_Float16)(o[dn][r] / lrun[r]);
    }
}

// ---------------------------------------------------------------------------
// Launch
// ---------------------------------------------------------------------------
extern "C" void kernel_launch(void* const* d_in, const int* in_sizes, int n_in,
                              void* d_out, int out_size, void* d_ws, size_t ws_size,
                              hipStream_t stream) {
  const float* x    = (const float*)d_in[0];
  const float* fr   = (const float*)d_in[1];
  // d_in[2] = mask (causal; computed analytically on device)
  const float* wq   = (const float*)d_in[3];
  const float* bq   = (const float*)d_in[4];
  const float* wk   = (const float*)d_in[5];
  const float* bk   = (const float*)d_in[6];
  const float* wv   = (const float*)d_in[7];
  const float* bv   = (const float*)d_in[8];
  const float* wo   = (const float*)d_in[9];
  const float* bo   = (const float*)d_in[10];
  float* out = (float*)d_out;

  const size_t BS = 4096;                 // B*S
  const size_t NQ = BS * 2048;
  const size_t NKV = BS * 512;

  char* ws = (char*)d_ws;
  size_t off = 0;
  auto alloc = [&](size_t bytes) -> char* {
    char* p = ws + off;
    off = (off + bytes + 255) & ~(size_t)255;
    return p;
  };
  _Float16* xh  = (_Float16*)alloc(NQ * 2);
  _Float16* wqh = (_Float16*)alloc((size_t)2048 * 2048 * 2);
  _Float16* wkh = (_Float16*)alloc((size_t)512 * 2048 * 2);
  _Float16* wvh = (_Float16*)alloc((size_t)512 * 2048 * 2);
  _Float16* woh = (_Float16*)alloc((size_t)2048 * 2048 * 2);
  float*    qf  = (float*)alloc(NQ * 4);
  float*    kf  = (float*)alloc(NKV * 4);
  float*    vf  = (float*)alloc(NKV * 4);
  _Float16* qh  = (_Float16*)alloc(NQ * 2);
  _Float16* khh = (_Float16*)alloc(NKV * 2);
  _Float16* vhh = (_Float16*)alloc(NKV * 2);
  _Float16* ah  = (_Float16*)alloc(NQ * 2);

  auto cvt = [&](const float* s, _Float16* d, size_t n) {
    size_t n4 = n / 4;
    cvt_f32_f16<<<(unsigned)((n4 + 255) / 256), 256, 0, stream>>>(s, d, n4);
  };
  cvt(x,  xh,  NQ);
  cvt(wq, wqh, (size_t)2048 * 2048);
  cvt(wk, wkh, (size_t)512 * 2048);
  cvt(wv, wvh, (size_t)512 * 2048);
  cvt(wo, woh, (size_t)2048 * 2048);

  // QKV projections
  gemm_f16f32<<<(4096 / BM) * (2048 / BN), 256, 0, stream>>>(xh, wqh, bq, qf, 4096, 2048, 2048);
  gemm_f16f32<<<(4096 / BM) * (512 / BN),  256, 0, stream>>>(xh, wkh, bk, kf, 4096, 512, 2048);
  gemm_f16f32<<<(4096 / BM) * (512 / BN),  256, 0, stream>>>(xh, wvh, bv, vf, 4096, 512, 2048);

  // RoPE + head-major f16 repack
  {
    size_t npq = NQ / 2;
    rope_pack<<<(unsigned)((npq + 255) / 256), 256, 0, stream>>>(qf, fr, qh, 32, 2048, npq);
    size_t npk = NKV / 2;
    rope_pack<<<(unsigned)((npk + 255) / 256), 256, 0, stream>>>(kf, fr, khh, 8, 512, npk);
    v_pack<<<(unsigned)((NKV + 255) / 256), 256, 0, stream>>>(vf, vhh, NKV);
  }

  // Flash attention: B * HEADS * (S/128) workgroups
  attn_fwd<<<2 * 32 * 16, 256, 0, stream>>>(qh, khh, vhh, ah);

  // Output projection -> f32 result
  gemm_f16f32<<<(4096 / BM) * (2048 / BN), 256, 0, stream>>>(ah, woh, bo, out, 4096, 2048, 2048);

  (void)in_sizes; (void)n_in; (void)out_size; (void)ws_size;
}
